// Projector_23278722744900
// MI455X (gfx1250) — compile-verified, matrix operand order checked by reading, not run on hardware
//
#include <hip/hip_runtime.h>

// ---------------------------------------------------------------------------
// MI455X (gfx1250): all matmul-shaped work through V_WMMA_F32_16X16X32_F16
// (wave32, f32 accumulate); GEMM tiles streamed to LDS with
// GLOBAL_LOAD_ASYNC_TO_LDS_B64 (ASYNCcnt) on the device path.
// ---------------------------------------------------------------------------

typedef __attribute__((ext_vector_type(16))) _Float16 v16h;
typedef __attribute__((ext_vector_type(8)))  float    v8f;

// --- CDNA5 async global->LDS copy (8 bytes / lane) -------------------------
// LDS address comes from the vdst VGPR; low 32 bits of a flat shared pointer are
// the LDS byte offset (ISA 10.2 aperture rules).  Fallback: load+store.
static __device__ __forceinline__ void async_copy_b64(_Float16* dst_lds,
                                                      const _Float16* src) {
#if defined(__gfx1250__)
  unsigned lds_off = (unsigned)(unsigned long long)dst_lds;
  unsigned long long ga = (unsigned long long)src;
  asm volatile("global_load_async_to_lds_b64 %0, %1, off"
               :: "v"(lds_off), "v"(ga)
               : "memory");
#else
  *(uint2*)dst_lds = *(const uint2*)src;
#endif
}

static __device__ __forceinline__ void async_wait0() {
#if defined(__gfx1250__)
  asm volatile("s_wait_asynccnt 0x0" ::: "memory");
#endif
}

// --- fragment gather from LDS ----------------------------------------------
// LDS tile layout: [row][32] f16, row-major.  Per CDNA5 ISA (16-bit A
// matrix 16x32): lane 0..15 holds row M=lane, K in {0..7,16..23}; lane 16..31
// holds K in {8..15,24..31}.  VGPR v packs K pair (kb,kb+1).
static __device__ __forceinline__ v16h load_frag(const _Float16* s, int row, int hi) {
  union { v16h v; unsigned int u[8]; } f;
  const unsigned int* p = (const unsigned int*)(s + row * 32);
#pragma unroll
  for (int v = 0; v < 8; ++v) {
    int kb = ((v >= 4) ? 16 : 0) + hi * 8 + (v & 3) * 2;
    f.u[v] = p[kb >> 1];
  }
  return f.v;
}

// ---------------------------------------------------------------------------
// Implicit-GEMM conv kernel.  C[m][n] = sum_k W[m][k] * im2col(act)[k][n]
//   K ordering: k = (ky*KHW + kx)*CIN + ci   (weights pre-packed this way)
//   act: f16 NHWC.  Block = 256 threads = 8 waves; tile 128(M) x 128(N);
//   wave = 64x32 => 4x2 accumulators => 8 wmma per K-step of 32.
// EPI: 0 = BN(scale,shift)+ReLU -> f16 NHWC
//      1 = +bias[m]             -> f16 NHWC
//      2 = +bias[m], guard m<M  -> f32 [n][ldC]   (text linear)
//      3 = +bias[z*bs+m]        -> f32 [z][m][H*W] (dynamic conv -> d_out)
// ---------------------------------------------------------------------------
template <int CIN, int KHW, int PAD, int EPI>
__global__ __launch_bounds__(256) void conv_wmma(
    const _Float16* __restrict__ act, const _Float16* __restrict__ wgt,
    void* __restrict__ outp, const float* __restrict__ p0,
    const float* __restrict__ p1, int H, int W, int M, int ldC,
    long actStride, long wStride, long outStride, int biasStride) {
  constexpr int KTOT = KHW * KHW * CIN;
  __shared__ _Float16 sA[128 * 32];
  __shared__ _Float16 sB[128 * 32];

  const int tid = threadIdx.x;
  const int z = blockIdx.z;
  act += (long)z * actStride;
  wgt += (long)z * wStride;

  const int n0 = blockIdx.x * 128;
  const int m0 = blockIdx.y * 128;
  const int lane = tid & 31, wave = tid >> 5;
  const int lane_lo = lane & 15, hi = lane >> 4;
  const int wm = (wave >> 2) * 64;   // wave M offset within 128 tile
  const int wn = (wave & 3) * 32;    // wave N offset within 128 tile
  const int HWp = H * W;

  // per-thread B-load assignment: pixel p, 16-channel half hh
  const int p = tid >> 1, hh = tid & 1;
  const int ng = n0 + p;
  const int bimg = ng / HWp;
  const int rem = ng - bimg * HWp;
  const int py = rem / W;
  const int px = rem - py * W;

  v8f acc[4][2];
#pragma unroll
  for (int mi = 0; mi < 4; ++mi)
#pragma unroll
    for (int ni = 0; ni < 2; ++ni)
#pragma unroll
      for (int r = 0; r < 8; ++r) acc[mi][ni][r] = 0.f;

  for (int k0 = 0; k0 < KTOT; k0 += 32) {
    int ky = 0, kx = 0, ci0 = k0;
    if (KHW > 1) {  // CIN % 32 == 0 => a 32-wide K step stays in one (ky,kx)
      int rr = k0 / CIN;
      ci0 = k0 - rr * CIN;
      ky = rr / KHW;
      kx = rr - (rr / KHW) * KHW;
    }
    // --- A tile: 128 rows x 32 halves, async 8B chunks, 4 per thread -------
#pragma unroll
    for (int j = 0; j < 4; ++j) {
      int ch = tid + j * 256;          // 0..1023
      int row = ch >> 3;
      int col = (ch & 7) * 4;
      _Float16* dst = &sA[row * 32 + col];
      if (m0 + row < M) {
        const _Float16* gp = wgt + (long)(m0 + row) * KTOT + k0 + col;
        async_copy_b64(dst, gp);
        if (j == 0 && k0 + 32 < KTOT)  // CDNA5 global_prefetch_b8 of next K-step
          __builtin_prefetch((const void*)(gp + 32), 0, 1);
      } else {
        uint2 zz = {0u, 0u};
        *(uint2*)dst = zz;
      }
    }
    // --- B tile: im2col gather, 32B contiguous NHWC per (pixel, half) ------
    {
      const int iy = py + ky - PAD;
      const int ix = px + kx - PAD;
      const bool ok = (PAD == 0) ||
                      ((unsigned)iy < (unsigned)H && (unsigned)ix < (unsigned)W);
      const _Float16* src =
          act + ((long)(bimg * H + iy) * W + ix) * CIN + ci0 + hh * 16;
#pragma unroll
      for (int j = 0; j < 4; ++j) {
        _Float16* dst = &sB[p * 32 + hh * 16 + j * 4];
        if (ok) {
          async_copy_b64(dst, src + j * 4);
        } else {
          uint2 zz = {0u, 0u};
          *(uint2*)dst = zz;
        }
      }
    }
    async_wait0();      // ASYNCcnt -> 0 : LDS tiles resident
    __syncthreads();

    v16h fa[4], fb[2];
#pragma unroll
    for (int mi = 0; mi < 4; ++mi)
      fa[mi] = load_frag(sA, wm + mi * 16 + lane_lo, hi);
#pragma unroll
    for (int ni = 0; ni < 2; ++ni)
      fb[ni] = load_frag(sB, wn + ni * 16 + lane_lo, hi);
#pragma unroll
    for (int mi = 0; mi < 4; ++mi)
#pragma unroll
      for (int ni = 0; ni < 2; ++ni)
        acc[mi][ni] = __builtin_amdgcn_wmma_f32_16x16x32_f16(
            false, fa[mi], false, fb[ni], (short)0, acc[mi][ni], false, false);
    __syncthreads();
  }

  // --- epilogue: C VGPR r holds M=r (lanes 0-15) / M=r+8 (lanes 16-31) -----
  if (EPI == 0 || EPI == 1) {
    // 8 consecutive m per (mi,hi) => one 16B f16 store per accumulator
#pragma unroll
    for (int mi = 0; mi < 4; ++mi)
#pragma unroll
      for (int ni = 0; ni < 2; ++ni) {
        const int mbase = m0 + wm + mi * 16 + hi * 8;
        const int n = n0 + wn + ni * 16 + lane_lo;
        union { uint4 q; _Float16 h[8]; } pack;
#pragma unroll
        for (int r = 0; r < 8; ++r) {
          const float a = acc[mi][ni][r];
          float vv = (EPI == 0) ? fmaxf(a * p0[mbase + r] + p1[mbase + r], 0.f)
                                : (a + p0[mbase + r]);
          pack.h[r] = (_Float16)vv;
        }
        *(uint4*)(&((_Float16*)outp)[(long)n * ldC + mbase]) = pack.q;
      }
  } else {
#pragma unroll
    for (int mi = 0; mi < 4; ++mi)
#pragma unroll
      for (int ni = 0; ni < 2; ++ni)
#pragma unroll
        for (int r = 0; r < 8; ++r) {
          const int m = m0 + wm + mi * 16 + hi * 8 + r;
          const int n = n0 + wn + ni * 16 + lane_lo;
          const float a = acc[mi][ni][r];
          if (EPI == 2) {
            if (m < M) ((float*)outp)[(long)n * ldC + m] = a + p0[m];
          } else {
            ((float*)outp)[(long)z * outStride + (long)m * HWp + n] =
                a + p0[z * biasStride + m];
          }
        }
  }
}

// ---------------------------------------------------------------------------
// Bilinear 2x upsample (half-pixel centers, matches jax.image.resize bilinear)
// NCHW_F32 ? fp32 NCHW input : f16 NHWC input.  Output f16 NHWC (B,2H,2W,C).
// ---------------------------------------------------------------------------
template <bool NCHW_F32>
__global__ void upsample2x(const void* __restrict__ in_, _Float16* __restrict__ out,
                           int Bn, int C, int H, int W) {
  const int OH = 2 * H, OW = 2 * W;
  const long total = (long)Bn * OH * OW * C;
  long idx = (long)blockIdx.x * blockDim.x + threadIdx.x;
  if (idx >= total) return;
  int c = (int)(idx % C);
  long t = idx / C;
  int ox = (int)(t % OW); t /= OW;
  int oy = (int)(t % OH);
  int b = (int)(t / OH);

  float fy = 0.5f * oy - 0.25f, fx = 0.5f * ox - 0.25f;
  int y0 = (int)floorf(fy), x0 = (int)floorf(fx);
  float wy = fy - (float)y0, wx = fx - (float)x0;
  int y0c = min(max(y0, 0), H - 1), y1c = min(y0 + 1, H - 1);
  int x0c = min(max(x0, 0), W - 1), x1c = min(x0 + 1, W - 1);

  float v00, v01, v10, v11;
  if (NCHW_F32) {
    const float* in = (const float*)in_;
    const float* base = in + ((long)b * C + c) * H * W;
    v00 = base[y0c * W + x0c]; v01 = base[y0c * W + x1c];
    v10 = base[y1c * W + x0c]; v11 = base[y1c * W + x1c];
  } else {
    const _Float16* in = (const _Float16*)in_;
    const _Float16* base = in + (long)b * H * W * C + c;
    v00 = (float)base[((long)y0c * W + x0c) * C];
    v01 = (float)base[((long)y0c * W + x1c) * C];
    v10 = (float)base[((long)y1c * W + x0c) * C];
    v11 = (float)base[((long)y1c * W + x1c) * C];
  }
  float v = (1.f - wy) * ((1.f - wx) * v00 + wx * v01) +
            wy * ((1.f - wx) * v10 + wx * v11);
  out[idx] = (_Float16)v;
}

// --- small prep kernels -----------------------------------------------------
__global__ void bn_prep(const float* g, const float* b, const float* m,
                        const float* v, float* scale, float* shift, int C) {
  int i = blockIdx.x * blockDim.x + threadIdx.x;
  if (i >= C) return;
  float s = g[i] * rsqrtf(v[i] + 1e-5f);
  scale[i] = s;
  shift[i] = b[i] - m[i] * s;
}

// OIHW fp32 -> [O][(ky*KW+kx)*I + ci] f16
__global__ void pack_w(const float* __restrict__ src, _Float16* __restrict__ dst,
                       int O, int I, int KK) {
  long d = (long)blockIdx.x * blockDim.x + threadIdx.x;
  long total = (long)O * I * KK;
  if (d >= total) return;
  int o = (int)(d / ((long)I * KK));
  int r = (int)(d - (long)o * I * KK);
  int kk = r / I;
  int ci = r - kk * I;
  dst[d] = (_Float16)src[((long)o * I + ci) * KK + kk];
}

__global__ void f32_to_f16(const float* __restrict__ s, _Float16* __restrict__ d,
                           long n) {
  long i = (long)blockIdx.x * blockDim.x + threadIdx.x;
  if (i < n) d[i] = (_Float16)s[i];
}

// t f32 [nc*16+b][2305] -> wk f16 [b][nc][(ky*3+kx)*256 + c]
__global__ void repack_wk(const float* __restrict__ t, _Float16* __restrict__ wk) {
  long d = (long)blockIdx.x * blockDim.x + threadIdx.x;
  const long total = 16L * 128 * 2304;
  if (d >= total) return;
  int b = (int)(d / (128L * 2304));
  int r = (int)(d - (long)b * 128 * 2304);
  int nc = r / 2304;
  int q = r - nc * 2304;
  int kk = q / 256;
  int c = q - kk * 256;
  wk[d] = (_Float16)t[(long)(nc * 16 + b) * 2305 + c * 9 + kk];
}

__global__ void extract_bias(const float* __restrict__ t, float* __restrict__ bias) {
  int i = blockIdx.x * blockDim.x + threadIdx.x;
  if (i >= 2048) return;
  int b = i / 128, nc = i - b * 128;
  bias[i] = t[(long)(nc * 16 + b) * 2305 + 2304];
}

// ---------------------------------------------------------------------------
static inline unsigned cdiv(long a, long b) { return (unsigned)((a + b - 1) / b); }

extern "C" void kernel_launch(void* const* d_in, const int* in_sizes, int n_in,
                              void* d_out, int out_size, void* d_ws, size_t ws_size,
                              hipStream_t stream) {
  const float* x      = (const float*)d_in[0];   // (16,512,24,24)
  const float* word   = (const float*)d_in[1];   // (128,16,768)
  const float* c1w    = (const float*)d_in[2];
  const float* bn1g   = (const float*)d_in[3];
  const float* bn1b   = (const float*)d_in[4];
  const float* bn1m   = (const float*)d_in[5];
  const float* bn1v   = (const float*)d_in[6];
  const float* c2w    = (const float*)d_in[7];
  const float* bn2g   = (const float*)d_in[8];
  const float* bn2b   = (const float*)d_in[9];
  const float* bn2m   = (const float*)d_in[10];
  const float* bn2v   = (const float*)d_in[11];
  const float* c3w    = (const float*)d_in[12];
  const float* c3b    = (const float*)d_in[13];
  const float* txtw   = (const float*)d_in[14];
  const float* txtb   = (const float*)d_in[15];

  // workspace bump allocator (256B aligned)
  char* ws = (char*)d_ws;
  size_t off = 0;
  auto alloc = [&](size_t bytes) -> void* {
    void* p = ws + off;
    off += (bytes + 255) & ~(size_t)255;
    return p;
  };
  _Float16* act0   = (_Float16*)alloc((size_t)16 * 96 * 96 * 512 * 2); // u1/u2/v
  _Float16* act1   = (_Float16*)alloc((size_t)16 * 96 * 96 * 256 * 2); // v1/v2
  _Float16* wA1    = (_Float16*)alloc((size_t)512 * 4608 * 2);
  _Float16* wA2    = (_Float16*)alloc((size_t)256 * 4608 * 2);
  _Float16* wA3    = (_Float16*)alloc((size_t)256 * 256 * 2);
  _Float16* txtw16 = (_Float16*)alloc((size_t)2305 * 768 * 2);
  _Float16* word16 = (_Float16*)alloc((size_t)2048 * 768 * 2);
  float*    tbuf   = (float*)alloc((size_t)2048 * 2305 * 4);
  _Float16* wkA    = (_Float16*)alloc((size_t)16 * 128 * 2304 * 2);
  float*    biasT  = (float*)alloc((size_t)2048 * 4);
  float*    sc1    = (float*)alloc(512 * 4);
  float*    sh1    = (float*)alloc(512 * 4);
  float*    sc2    = (float*)alloc(256 * 4);
  float*    sh2    = (float*)alloc(256 * 4);

  const int TB = 256;

  // --- prep: BN fold, weight/word packing to f16 ---------------------------
  bn_prep<<<cdiv(512, TB), TB, 0, stream>>>(bn1g, bn1b, bn1m, bn1v, sc1, sh1, 512);
  bn_prep<<<cdiv(256, TB), TB, 0, stream>>>(bn2g, bn2b, bn2m, bn2v, sc2, sh2, 256);
  pack_w<<<cdiv(512L * 512 * 9, TB), TB, 0, stream>>>(c1w, wA1, 512, 512, 9);
  pack_w<<<cdiv(256L * 512 * 9, TB), TB, 0, stream>>>(c2w, wA2, 256, 512, 9);
  pack_w<<<cdiv(256L * 256, TB), TB, 0, stream>>>(c3w, wA3, 256, 256, 1);
  f32_to_f16<<<cdiv(2305L * 768, TB), TB, 0, stream>>>(txtw, txtw16, 2305L * 768);
  f32_to_f16<<<cdiv(2048L * 768, TB), TB, 0, stream>>>(word, word16, 2048L * 768);

  // --- text branch: t = word @ txt_w.T + txt_b  (M=2305, N=2048, K=768) ----
  conv_wmma<768, 1, 0, 2><<<dim3(16, 19, 1), TB, 0, stream>>>(
      word16, txtw16, tbuf, txtb, nullptr, 1, 2048, 2305, 2305, 0, 0, 0, 0);
  repack_wk<<<cdiv(16L * 128 * 2304, TB), TB, 0, stream>>>(tbuf, wkA);
  extract_bias<<<cdiv(2048, TB), TB, 0, stream>>>(tbuf, biasT);

  // --- vision backbone ------------------------------------------------------
  // u1 = up2(x): (16,48,48,512) f16 NHWC
  upsample2x<true><<<cdiv(16L * 48 * 48 * 512, TB), TB, 0, stream>>>(
      x, act0, 16, 512, 24, 24);
  // v1 = relu(bn1(conv1(u1))): M=512, N=16*48*48, K=4608
  conv_wmma<512, 3, 1, 0><<<dim3(288, 4, 1), TB, 0, stream>>>(
      act0, wA1, act1, sc1, sh1, 48, 48, 512, 512, 0, 0, 0, 0);
  // u2 = up2(v1): (16,96,96,512)
  upsample2x<false><<<cdiv(16L * 96 * 96 * 512, TB), TB, 0, stream>>>(
      act1, act0, 16, 512, 48, 48);
  // v2 = relu(bn2(conv2(u2))): M=256, N=16*96*96, K=4608
  conv_wmma<512, 3, 1, 0><<<dim3(1152, 2, 1), TB, 0, stream>>>(
      act0, wA2, act1, sc2, sh2, 96, 96, 256, 256, 0, 0, 0, 0);
  // v = conv3_1x1(v2) + b: M=256, K=256
  conv_wmma<256, 1, 0, 1><<<dim3(1152, 2, 1), TB, 0, stream>>>(
      act1, wA3, act0, c3b, nullptr, 96, 96, 256, 256, 0, 0, 0, 0);

  // --- dynamic per-(image,word) conv: per b: M=128(nc), N=9216, K=2304 -----
  conv_wmma<256, 3, 1, 3><<<dim3(72, 1, 16), TB, 0, stream>>>(
      act0, wkA, d_out, biasT, nullptr, 96, 96, 128, 0,
      (long)96 * 96 * 256, (long)128 * 2304, (long)128 * 9216, 128);

  (void)in_sizes; (void)n_in; (void)out_size; (void)ws_size;
}
